// MSABlock_84817014161973
// MI455X (gfx1250) — compile-verified, matrix operand order checked by reading, not run on hardware
//
#include <hip/hip_runtime.h>

// ---------------------------------------------------------------------------
// MSABlock for MI455X (gfx1250). All contractions run on
// v_wmma_f32_16x16x32_bf16 (bf16 operands, f32 accum). Memory-bound block:
// ~160 GFLOP vs ~1.5GB traffic @ 23.3 TB/s -> keep operands bf16, fuse
// residual adds into GEMM epilogues, chunk giant intermediates, stage tiles
// through LDS with b128 loads + global prefetch of the next K tile.
// ---------------------------------------------------------------------------

typedef unsigned short u16;
typedef __attribute__((ext_vector_type(16))) __bf16        v16bf;
typedef __attribute__((ext_vector_type(8)))  float         v8f;
typedef __attribute__((ext_vector_type(4)))  unsigned int  u32x4;

#define S_MSA 512
#define NTOK  256
#define CM    64
#define CZ    128

__device__ __forceinline__ u16 f2bfu(float f) {
    union { float f; unsigned u; } x; x.f = f;
    unsigned r = x.u + 0x7fffu + ((x.u >> 16) & 1u);
    return (u16)(r >> 16);
}
__device__ __forceinline__ float bf2f(u16 h) {
    union { float f; unsigned u; } x; x.u = ((unsigned)h) << 16;
    return x.f;
}
__device__ __forceinline__ float sigm(float x) { return 1.f / (1.f + __expf(-x)); }

// ---------------------------------------------------------------------------
// Generic batched NT GEMM:  C[b1,b2] = alpha * A[b1,b2] (MxK) * B[b1,b2] (NxK)^T
// A element (m,k)  at  A + b1*bsA1 + b2*bsA2 + m*rsA + k*csA   (bf16)
// B element (n,k)  at  B + b1*bsB1 + b2*bsB2 + n*rsB + k*csB   (bf16)
// C element (m,n)  at  C + b1*bsC1 + b2*bsC2 + (m/cM2)*csHi + (m%cM2)*rsC + n*csC
// flags: 1 = write bf16, 2 = accumulate into f32 C, 0 = write f32.
// K must be a multiple of 32 (true for every call here).
// When cs==1 the operand's rows are K-contiguous and 16B aligned (all call
// sites guarantee rs and base offsets are multiples of 8 elements) -> tiles
// are staged with 128-bit loads; otherwise a scalar gather path is used.
// ---------------------------------------------------------------------------
struct GemmArgs {
    const u16* A; const u16* B;
    float* Cf; u16* Cb;
    int M, N, K;
    long rsA, csA, bsA1, bsA2;
    long rsB, csB, bsB1, bsB2;
    long rsC, csC, bsC1, bsC2, csHi;
    int cM2, nb2, flags;
    float alpha;
};

union Frag { v16bf v; u32x4 q[2]; };

#define BM 64
#define BN 64
#define BK 32

__global__ __launch_bounds__(128) void msab_gemm(GemmArgs g) {
    __shared__ __align__(16) u16 As[BM * 40];
    __shared__ __align__(16) u16 Bs[BN * 40];

    int bz = blockIdx.z;
    int b2 = (g.nb2 > 1) ? (bz % g.nb2) : 0;
    int b1 = (g.nb2 > 1) ? (bz / g.nb2) : bz;
    const u16* A = g.A + (long)b1 * g.bsA1 + (long)b2 * g.bsA2;
    const u16* B = g.B + (long)b1 * g.bsB1 + (long)b2 * g.bsB2;
    long cbase = (long)b1 * g.bsC1 + (long)b2 * g.bsC2;

    int m0 = blockIdx.x * BM, n0 = blockIdx.y * BN;
    int tid = threadIdx.x;
    int wave = tid >> 5, lane = tid & 31;
    int wm = (wave >> 1) * 32, wn = (wave & 1) * 32;

    const u32x4 qzero = {0u, 0u, 0u, 0u};
    v8f vzero = {0.f,0.f,0.f,0.f,0.f,0.f,0.f,0.f};
    v8f acc[2][2];
    for (int i = 0; i < 2; ++i) for (int j = 0; j < 2; ++j) acc[i][j] = vzero;

    const bool fastA = (g.csA == 1);
    const bool fastB = (g.csB == 1);

    for (int k0 = 0; k0 < g.K; k0 += BK) {
        // ---- stage A tile ----
        if (fastA) {
            #pragma unroll
            for (int it = 0; it < 2; ++it) {
                int idx = tid + it * 128;           // 256 chunks of 8 halfs
                int mm = idx >> 2, kc = (idx & 3) * 8;
                int m = m0 + mm;
                u32x4 val = qzero;
                if (m < g.M) {
                    const u16* src = A + (long)m * g.rsA + (long)(k0 + kc);
                    val = *reinterpret_cast<const u32x4*>(src);
                    if (k0 + BK < g.K)
                        __builtin_prefetch(src + BK, 0, 1);   // global_prefetch_b8
                }
                *reinterpret_cast<u32x4*>(&As[mm * 40 + kc]) = val;
            }
        } else {
            for (int idx = tid; idx < BM * BK; idx += 128) {
                int mm = idx >> 5, kk = idx & 31;
                int m = m0 + mm;
                u16 va = 0;
                if (m < g.M) va = A[(long)m * g.rsA + (long)(k0 + kk) * g.csA];
                As[mm * 40 + kk] = va;
            }
        }
        // ---- stage B tile ----
        if (fastB) {
            #pragma unroll
            for (int it = 0; it < 2; ++it) {
                int idx = tid + it * 128;
                int mm = idx >> 2, kc = (idx & 3) * 8;
                int n = n0 + mm;
                u32x4 val = qzero;
                if (n < g.N) {
                    const u16* src = B + (long)n * g.rsB + (long)(k0 + kc);
                    val = *reinterpret_cast<const u32x4*>(src);
                    if (k0 + BK < g.K)
                        __builtin_prefetch(src + BK, 0, 1);
                }
                *reinterpret_cast<u32x4*>(&Bs[mm * 40 + kc]) = val;
            }
        } else {
            for (int idx = tid; idx < BM * BK; idx += 128) {
                int mm = idx >> 5, kk = idx & 31;
                int n = n0 + mm;
                u16 vb = 0;
                if (n < g.N) vb = B[(long)n * g.rsB + (long)(k0 + kk) * g.csB];
                Bs[mm * 40 + kk] = vb;
            }
        }
        __syncthreads();

        // ---- fragments (ISA 7.12.2: lane row = lane&15, K split by lane half)
        int row = lane & 15, kb = (lane >> 4) * 8;
        Frag af[2], bfr[2];
        #pragma unroll
        for (int t = 0; t < 2; ++t) {
            const u16* pa = &As[(wm + t * 16 + row) * 40 + kb];
            af[t].q[0] = *reinterpret_cast<const u32x4*>(pa);
            af[t].q[1] = *reinterpret_cast<const u32x4*>(pa + 16);
            const u16* pb = &Bs[(wn + t * 16 + row) * 40 + kb];
            bfr[t].q[0] = *reinterpret_cast<const u32x4*>(pb);
            bfr[t].q[1] = *reinterpret_cast<const u32x4*>(pb + 16);
        }
        #pragma unroll
        for (int i = 0; i < 2; ++i)
            #pragma unroll
            for (int j = 0; j < 2; ++j)
                acc[i][j] = __builtin_amdgcn_wmma_f32_16x16x32_bf16(
                    false, af[i].v, false, bfr[j].v, (short)0, acc[i][j], false, false);
        __syncthreads();
    }

    #pragma unroll
    for (int i = 0; i < 2; ++i) {
        #pragma unroll
        for (int j = 0; j < 2; ++j) {
            int nn = n0 + wn + j * 16 + (lane & 15);
            if (nn >= g.N) continue;
            #pragma unroll
            for (int r = 0; r < 8; ++r) {
                int mm = m0 + wm + i * 16 + ((lane >> 4) * 8) + r;
                if (mm >= g.M) continue;
                long off = cbase + (long)(mm / g.cM2) * g.csHi
                                 + (long)(mm % g.cM2) * g.rsC + (long)nn * g.csC;
                float v = g.alpha * acc[i][j][r];
                if (g.flags & 1)      g.Cb[off] = f2bfu(v);
                else if (g.flags & 2) g.Cf[off] += v;
                else                  g.Cf[off] = v;
            }
        }
    }
}

// --------------------------- elementwise kernels ---------------------------

__global__ void msab_copy(const float* a, float* b, long n) {
    long i = (long)blockIdx.x * blockDim.x + threadIdx.x;
    if (i < n) b[i] = a[i];
}

// Wt[n,k] = bf16(W[k,n])
__global__ void msab_wconv(const float* W, u16* Wt, int K, int N) {
    int i = blockIdx.x * blockDim.x + threadIdx.x;
    if (i >= K * N) return;
    int n = i / K, k = i % K;
    Wt[i] = f2bfu(W[(long)k * N + n]);
}

// wave-per-row LayerNorm, bf16 out. src offset = srow*rstride + c*cstride.
// trans!=0: srow = (r%R1)*R0 + r/R1  (used for the swapaxes attention pass)
__global__ void msab_ln(const float* x, const float* gam, const float* bet,
                        u16* out, int R, int C, long rstride, long cstride,
                        int trans, int R0, int R1) {
    int r = blockIdx.x * (blockDim.x >> 5) + (threadIdx.x >> 5);
    if (r >= R) return;
    int lane = threadIdx.x & 31;
    long srow = trans ? ((long)(r % R1) * R0 + (long)(r / R1)) : (long)r;
    float s = 0.f, ss = 0.f;
    for (int c = lane; c < C; c += 32) {
        float v = x[srow * rstride + (long)c * cstride];
        s += v; ss += v * v;
    }
    for (int off = 16; off > 0; off >>= 1) {
        s  += __shfl_xor(s,  off, 32);
        ss += __shfl_xor(ss, off, 32);
    }
    float mu = s / C, var = ss / C - mu * mu;
    float inv = rsqrtf(var + 1e-5f);
    for (int c = lane; c < C; c += 32) {
        float v = (x[srow * rstride + (long)c * cstride] - mu) * inv * gam[c] + bet[c];
        out[(long)r * C + c] = f2bfu(v);
    }
}

// out = bf16( silu(a) * b )
__global__ void msab_silu_mul(const u16* a, const u16* b, u16* o, long n) {
    long i = (long)blockIdx.x * blockDim.x + threadIdx.x;
    if (i >= n) return;
    float x = bf2f(a[i]);
    o[i] = f2bfu(x * sigm(x) * bf2f(b[i]));
}

// tri-mul gate: dst[c][p][q] (or [c][q][p]) = bf16(mask[p,q]*sigmoid(gate)*proj)
__global__ void msab_gate_perm(const float* proj, const float* gate,
                               const float* mask, u16* out, int trans) {
    long i = (long)blockIdx.x * blockDim.x + threadIdx.x;
    if (i >= (long)NTOK * NTOK * 128) return;
    int c = (int)(i & 127);
    long r = i >> 7;
    int p = (int)(r >> 8), q = (int)(r & 255);
    float v = mask[r] * sigm(gate[i]) * proj[i];
    long dst = (long)c * (NTOK * NTOK) + (trans ? (q * NTOK + p) : (p * NTOK + q));
    out[dst] = f2bfu(v);
}

// z += sigmoid(g) * t
__global__ void msab_sig_gate_add(const float* g, const float* t, float* z, long n) {
    long i = (long)blockIdx.x * blockDim.x + threadIdx.x;
    if (i < n) z[i] += sigm(g[i]) * t[i];
}

// PWA: softmax over j of b[i,j,h]  (layout [(i,j),h], h=8)
__global__ void msab_pwa_softmax(const float* b, float* w) {
    int ih = blockIdx.x * blockDim.x + threadIdx.x;
    if (ih >= NTOK * 8) return;
    int i = ih >> 3, h = ih & 7;
    const float* p = b + (long)i * (NTOK * 8) + h;
    float mx = -3.0e38f;
    for (int j = 0; j < NTOK; ++j) { float v = p[(long)j * 8]; mx = v > mx ? v : mx; }
    float s = 0.f;
    for (int j = 0; j < NTOK; ++j) s += __expf(p[(long)j * 8] - mx);
    float inv = 1.f / s;
    float* wp = w + (long)i * (NTOK * 8) + h;
    for (int j = 0; j < NTOK; ++j) wp[(long)j * 8] = __expf(p[(long)j * 8] - mx) * inv;
}

// PWA: o[s,i,h,c] = bf16( sigmoid(g[s,i,hc]) * sum_j w[i,j,h]*v[s,j,h,c] )
__global__ void msab_pwa_wv(const u16* v, const float* w, const float* g, u16* o) {
    long idx = (long)blockIdx.x * blockDim.x + threadIdx.x;
    if (idx >= (long)S_MSA * NTOK * 64) return;
    int c = (int)(idx & 7);
    int h = (int)((idx >> 3) & 7);
    int i = (int)((idx >> 6) & 255);
    long s = idx >> 14;
    const u16* vp = v + s * (NTOK * 64) + h * 8 + c;
    const float* wp = w + (long)i * (NTOK * 8) + h;
    float acc = 0.f;
    for (int j = 0; j < NTOK; ++j) acc += wp[(long)j * 8] * bf2f(vp[(long)j * 64]);
    o[idx] = f2bfu(sigm(g[idx]) * acc);
}

// triangle attention softmax, logits chunk [32][4][256][256] f32:
// p = softmax_k( logits*scale + tb[(b,k),h] + (mask[a*ms0+k*ms1]-1)*1e9 )
__global__ void msab_tri_softmax(const float* logits, const float* tb,
                                 const float* mask, u16* p, int a0,
                                 long ms0, long ms1, float scale) {
    int row = blockIdx.x * blockDim.x + threadIdx.x;
    if (row >= 32 * 4 * 256) return;
    int b = row & 255, h = (row >> 8) & 3;
    int a = a0 + (row >> 10);
    const float* lp = logits + (long)row * 256;
    float mx = -3.0e38f;
    for (int k = 0; k < 256; ++k) {
        float v = lp[k] * scale + tb[((long)b * 256 + k) * 4 + h]
                + (mask[(long)a * ms0 + (long)k * ms1] - 1.f) * 1e9f;
        mx = v > mx ? v : mx;
    }
    float s = 0.f;
    for (int k = 0; k < 256; ++k) {
        float v = lp[k] * scale + tb[((long)b * 256 + k) * 4 + h]
                + (mask[(long)a * ms0 + (long)k * ms1] - 1.f) * 1e9f;
        s += __expf(v - mx);
    }
    float inv = 1.f / s;
    u16* pp = p + (long)row * 256;
    for (int k = 0; k < 256; ++k) {
        float v = lp[k] * scale + tb[((long)b * 256 + k) * 4 + h]
                + (mask[(long)a * ms0 + (long)k * ms1] - 1.f) * 1e9f;
        pp[k] = f2bfu(__expf(v - mx) * inv);
    }
}

// attention gate: og[(a_l,b),hc] = bf16( sigmoid(g[(a0+a_l,b),hc]) * o )
__global__ void msab_att_gate(const float* g, const float* o, u16* og, int a0) {
    long idx = (long)blockIdx.x * blockDim.x + threadIdx.x;
    if (idx >= (long)32 * 256 * 128) return;
    int hc = (int)(idx & 127);
    long ab = idx >> 7;
    long gi = ((long)a0 * 256 + ab) * 128 + hc;
    og[idx] = f2bfu(sigm(g[gi]) * o[idx]);
}

// --------------------------- host side -------------------------------------

static void gemmL(hipStream_t st, const u16* A, const u16* B, void* C,
                  int M, int N, int K,
                  long rsA, long csA, long bsA1, long bsA2,
                  long rsB, long csB, long bsB1, long bsB2,
                  long rsC, long csC, long bsC1, long bsC2,
                  int cM2, long csHi, int nb1, int nb2, float alpha, int flags) {
    GemmArgs g;
    g.A = A; g.B = B; g.Cf = (float*)C; g.Cb = (u16*)C;
    g.M = M; g.N = N; g.K = K;
    g.rsA = rsA; g.csA = csA; g.bsA1 = bsA1; g.bsA2 = bsA2;
    g.rsB = rsB; g.csB = csB; g.bsB1 = bsB1; g.bsB2 = bsB2;
    g.rsC = rsC; g.csC = csC; g.bsC1 = bsC1; g.bsC2 = bsC2;
    g.cM2 = cM2; g.csHi = csHi; g.nb2 = nb2; g.flags = flags; g.alpha = alpha;
    dim3 grid((M + BM - 1) / BM, (N + BN - 1) / BN, nb1 * nb2);
    msab_gemm<<<grid, dim3(128), 0, st>>>(g);
}

// input indices: m,z,mask then params pytree leaves (dict keys sorted)
enum {
    IN_M = 0, IN_Z, IN_MASK,
    MT_WA, MT_WB, MT_WO, MT_LNB, MT_LNG,
    OPM_WA, OPM_WB, OPM_WO, OPM_LNB, OPM_LNG,
    PT_WA, PT_WB, PT_WO, PT_LNB, PT_LNG,
    PWA_WB, PWA_WG, PWA_WO, PWA_WV, PWA_LNMB, PWA_LNMG, PWA_LNZB, PWA_LNZG,
    TAE_WBIAS, TAE_WG, TAE_WK, TAE_WO, TAE_WQ, TAE_WV, TAE_LNB, TAE_LNG,
    TAS_WBIAS, TAS_WG, TAS_WK, TAS_WO, TAS_WQ, TAS_WV, TAS_LNB, TAS_LNG,
    TMI_WAG, TMI_WAP, TMI_WBG, TMI_WBP, TMI_WG, TMI_WO,
    TMI_LNIB, TMI_LNIG, TMI_LNOB, TMI_LNOG,
    TMO_WAG, TMO_WAP, TMO_WBG, TMO_WBP, TMO_WG, TMO_WO,
    TMO_LNIB, TMO_LNIG, TMO_LNOB, TMO_LNOG,
};

extern "C" void kernel_launch(void* const* d_in, const int* in_sizes, int n_in,
                              void* d_out, int out_size, void* d_ws, size_t ws_size,
                              hipStream_t stream) {
    (void)in_sizes; (void)n_in; (void)out_size; (void)ws_size;

    const float* in_m  = (const float*)d_in[IN_M];
    const float* in_z  = (const float*)d_in[IN_Z];
    const float* mask  = (const float*)d_in[IN_MASK];
    float* out_m = (float*)d_out;
    float* out_z = out_m + (long)S_MSA * NTOK * CM;

    char* ws = (char*)d_ws;
    const size_t MB = 1u << 20;
    // arenas (~197 MB total)
    size_t oWT = 0;
    size_t oA0 = 2 * MB;         // 17 MB  bf16 LN / softmax-prob buffers
    size_t oA1 = oA0 + 17 * MB;  // 34 MB
    size_t oA2 = oA1 + 34 * MB;  // 34 MB
    size_t oA3 = oA2 + 34 * MB;  // 34 MB
    size_t oA4 = oA3 + 34 * MB;  // 34 MB
    size_t oA5 = oA4 + 34 * MB;  // 34 MB
    size_t oA6 = oA5 + 34 * MB;  // 8 MB small buffers
    auto F = [&](size_t o) { return (float*)(ws + o); };
    auto H = [&](size_t o) { return (u16*)(ws + o); };

    const int BIG = 1 << 30;
    const long Z0 = 0;

    // ---- init residual streams in d_out ----
    {
        long n = (long)S_MSA * NTOK * CM;
        msab_copy<<<dim3((unsigned)((n + 255) / 256)), dim3(256), 0, stream>>>(in_m, out_m, n);
        long nz = (long)NTOK * NTOK * CZ;
        msab_copy<<<dim3((unsigned)((nz + 255) / 256)), dim3(256), 0, stream>>>(in_z, out_z, nz);
    }

    // ---- convert all weights to transposed bf16 [N,K] ----
    size_t woff = 0;
    auto cvt = [&](int idx, int K, int N) -> u16* {
        u16* dst = H(oWT) + woff; woff += (size_t)K * N;
        int tot = K * N;
        msab_wconv<<<dim3((tot + 255) / 256), dim3(256), 0, stream>>>(
            (const float*)d_in[idx], dst, K, N);
        return dst;
    };
    u16 *wMTa = cvt(MT_WA, 64, 256), *wMTb = cvt(MT_WB, 64, 256), *wMTo = cvt(MT_WO, 256, 64);
    u16 *wOPa = cvt(OPM_WA, 64, 32), *wOPb = cvt(OPM_WB, 64, 32), *wOPo = cvt(OPM_WO, 1024, 128);
    u16 *wPTa = cvt(PT_WA, 128, 512), *wPTb = cvt(PT_WB, 128, 512), *wPTo = cvt(PT_WO, 512, 128);
    u16 *wPWb = cvt(PWA_WB, 128, 8), *wPWg = cvt(PWA_WG, 64, 64),
        *wPWo = cvt(PWA_WO, 64, 64), *wPWv = cvt(PWA_WV, 64, 64);
    u16 *wTA_bias[2], *wTA_g[2], *wTA_k[2], *wTA_o[2], *wTA_q[2], *wTA_v[2];
    // pass 0 = start, pass 1 = end
    wTA_bias[1] = cvt(TAE_WBIAS, 128, 4); wTA_g[1] = cvt(TAE_WG, 128, 128);
    wTA_k[1] = cvt(TAE_WK, 128, 128);     wTA_o[1] = cvt(TAE_WO, 128, 128);
    wTA_q[1] = cvt(TAE_WQ, 128, 128);     wTA_v[1] = cvt(TAE_WV, 128, 128);
    wTA_bias[0] = cvt(TAS_WBIAS, 128, 4); wTA_g[0] = cvt(TAS_WG, 128, 128);
    wTA_k[0] = cvt(TAS_WK, 128, 128);     wTA_o[0] = cvt(TAS_WO, 128, 128);
    wTA_q[0] = cvt(TAS_WQ, 128, 128);     wTA_v[0] = cvt(TAS_WV, 128, 128);
    u16 *wTM_ag[2], *wTM_ap[2], *wTM_bg[2], *wTM_bp[2], *wTM_g[2], *wTM_o[2];
    // pass 0 = outgoing (tri_mul_out), pass 1 = incoming (tri_mul_in)
    wTM_ag[1] = cvt(TMI_WAG, 128, 128); wTM_ap[1] = cvt(TMI_WAP, 128, 128);
    wTM_bg[1] = cvt(TMI_WBG, 128, 128); wTM_bp[1] = cvt(TMI_WBP, 128, 128);
    wTM_g[1]  = cvt(TMI_WG, 128, 128);  wTM_o[1]  = cvt(TMI_WO, 128, 128);
    wTM_ag[0] = cvt(TMO_WAG, 128, 128); wTM_ap[0] = cvt(TMO_WAP, 128, 128);
    wTM_bg[0] = cvt(TMO_WBG, 128, 128); wTM_bp[0] = cvt(TMO_WBP, 128, 128);
    wTM_g[0]  = cvt(TMO_WG, 128, 128);  wTM_o[0]  = cvt(TMO_WO, 128, 128);

    auto ln = [&](const float* x, int gi, int bi, u16* out, int R, int C,
                  long rstride, long cstride, int trans) {
        dim3 grid((R + 7) / 8);
        msab_ln<<<grid, dim3(256), 0, stream>>>(
            x, (const float*)d_in[gi], (const float*)d_in[bi],
            out, R, C, rstride, cstride, trans, NTOK, NTOK);
    };

    // =============== OuterProductMean: z += opm(m) ===============
    {
        ln(out_m, OPM_LNG, OPM_LNB, H(oA0), S_MSA * NTOK, CM, CM, 1, 0);
        // a,b bf16 [s,(i,c)] = [512,8192]
        gemmL(stream, H(oA0), wOPa, H(oA1), S_MSA * NTOK, 32, 64,
              64, 1, Z0, Z0, 64, 1, Z0, Z0, 32, 1, Z0, Z0, BIG, 0, 1, 1, 1.f, 1);
        gemmL(stream, H(oA0), wOPb, H(oA2), S_MSA * NTOK, 32, 64,
              64, 1, Z0, Z0, 64, 1, Z0, Z0, 32, 1, Z0, Z0, BIG, 0, 1, 1, 1.f, 1);
        for (int i0 = 0; i0 < NTOK; i0 += 64) {
            // G chunk bf16 [i_l][c][ (j,d) ]  (batch over i_l)
            gemmL(stream, H(oA1) + (size_t)i0 * 32, H(oA2), H(oA3),
                  32, 8192, 512,
                  1, 8192, 32, Z0,
                  1, 8192, Z0, Z0,
                  8192, 1, 262144, Z0, BIG, 0, 64, 1, 1.f / 512.f, 1);
            // project per c: z[(i0+i_l),j,:] += G_chunk[i_l][c][j,:d] @ Wo[c*32+d, :]
            for (int c = 0; c < 32; ++c) {
                gemmL(stream, H(oA3) + (size_t)c * 8192, wOPo + (size_t)c * 32,
                      out_z + (size_t)i0 * NTOK * CZ,
                      256, 128, 32,
                      32, 1, 262144, Z0,
                      1024, 1, Z0, Z0,
                      128, 1, 32768, Z0, BIG, 0, 64, 1, 1.f, 2);
            }
        }
    }

    // =============== MSA pair weighted averaging: m += pwa(m,z) ===============
    {
        ln(out_m, PWA_LNMG, PWA_LNMB, H(oA0), S_MSA * NTOK, CM, CM, 1, 0);
        gemmL(stream, H(oA0), wPWv, H(oA3), S_MSA * NTOK, 64, 64,
              64, 1, Z0, Z0, 64, 1, Z0, Z0, 64, 1, Z0, Z0, BIG, 0, 1, 1, 1.f, 1);
        gemmL(stream, H(oA0), wPWg, F(oA1), S_MSA * NTOK, 64, 64,
              64, 1, Z0, Z0, 64, 1, Z0, Z0, 64, 1, Z0, Z0, BIG, 0, 1, 1, 1.f, 0);
        ln(out_z, PWA_LNZG, PWA_LNZB, H(oA0), NTOK * NTOK, CZ, CZ, 1, 0);
        gemmL(stream, H(oA0), wPWb, F(oA6), NTOK * NTOK, 8, 128,
              128, 1, Z0, Z0, 128, 1, Z0, Z0, 8, 1, Z0, Z0, BIG, 0, 1, 1, 1.f, 0);
        msab_pwa_softmax<<<dim3(8), dim3(256), 0, stream>>>(F(oA6), F(oA6 + 2 * MB));
        {
            long n = (long)S_MSA * NTOK * 64;
            msab_pwa_wv<<<dim3((unsigned)((n + 255) / 256)), dim3(256), 0, stream>>>(
                H(oA3), F(oA6 + 2 * MB), F(oA1), H(oA4));
        }
        gemmL(stream, H(oA4), wPWo, out_m, S_MSA * NTOK, 64, 64,
              64, 1, Z0, Z0, 64, 1, Z0, Z0, 64, 1, Z0, Z0, BIG, 0, 1, 1, 1.f, 2);
    }

    // =============== MSA transition (SwiGLU, n=4), chunked ===============
    for (int r0 = 0; r0 < S_MSA * NTOK; r0 += 32768) {
        float* xin = out_m + (size_t)r0 * CM;
        ln(xin, MT_LNG, MT_LNB, H(oA0), 32768, CM, CM, 1, 0);
        gemmL(stream, H(oA0), wMTa, H(oA3), 32768, 256, 64,
              64, 1, Z0, Z0, 64, 1, Z0, Z0, 256, 1, Z0, Z0, BIG, 0, 1, 1, 1.f, 1);
        gemmL(stream, H(oA0), wMTb, H(oA4), 32768, 256, 64,
              64, 1, Z0, Z0, 64, 1, Z0, Z0, 256, 1, Z0, Z0, BIG, 0, 1, 1, 1.f, 1);
        long n = 32768L * 256;
        msab_silu_mul<<<dim3((unsigned)((n + 255) / 256)), dim3(256), 0, stream>>>(
            H(oA3), H(oA4), H(oA5), n);
        gemmL(stream, H(oA5), wMTo, xin, 32768, 64, 256,
              256, 1, Z0, Z0, 256, 1, Z0, Z0, 64, 1, Z0, Z0, BIG, 0, 1, 1, 1.f, 2);
    }

    // =============== triangle multiplicative updates ===============
    for (int pass = 0; pass < 2; ++pass) {  // 0 = outgoing, 1 = incoming
        int lnig = pass ? TMI_LNIG : TMO_LNIG, lnib = pass ? TMI_LNIB : TMO_LNIB;
        int lnog = pass ? TMI_LNOG : TMO_LNOG, lnob = pass ? TMI_LNOB : TMO_LNOB;
        ln(out_z, lnig, lnib, H(oA0), NTOK * NTOK, CZ, CZ, 1, 0);
        long n8 = (long)NTOK * NTOK * 128;
        // a = mask * sigmoid(h Wag) * (h Wap), stored [c][.][.] bf16
        gemmL(stream, H(oA0), wTM_ap[pass], F(oA1), NTOK * NTOK, 128, 128,
              128, 1, Z0, Z0, 128, 1, Z0, Z0, 128, 1, Z0, Z0, BIG, 0, 1, 1, 1.f, 0);
        gemmL(stream, H(oA0), wTM_ag[pass], F(oA2), NTOK * NTOK, 128, 128,
              128, 1, Z0, Z0, 128, 1, Z0, Z0, 128, 1, Z0, Z0, BIG, 0, 1, 1, 1.f, 0);
        msab_gate_perm<<<dim3((unsigned)((n8 + 255) / 256)), dim3(256), 0, stream>>>(
            F(oA1), F(oA2), mask, H(oA3), pass);
        gemmL(stream, H(oA0), wTM_bp[pass], F(oA1), NTOK * NTOK, 128, 128,
              128, 1, Z0, Z0, 128, 1, Z0, Z0, 128, 1, Z0, Z0, BIG, 0, 1, 1, 1.f, 0);
        gemmL(stream, H(oA0), wTM_bg[pass], F(oA2), NTOK * NTOK, 128, 128,
              128, 1, Z0, Z0, 128, 1, Z0, Z0, 128, 1, Z0, Z0, BIG, 0, 1, 1, 1.f, 0);
        msab_gate_perm<<<dim3((unsigned)((n8 + 255) / 256)), dim3(256), 0, stream>>>(
            F(oA1), F(oA2), mask, H(oA4), pass);
        // x[c][i][j] = sum_k a_c[i,k] b_c[j,k]  (batched NT over c=128)
        gemmL(stream, H(oA3), H(oA4), F(oA1), 256, 256, 256,
              256, 1, 65536, Z0, 256, 1, 65536, Z0,
              256, 1, 65536, Z0, BIG, 0, 128, 1, 1.f, 0);
        // hx = LN_out(x) with channel stride 65536
        ln(F(oA1), lnog, lnob, H(oA5), NTOK * NTOK, 128, 1, 65536, 0);
        gemmL(stream, H(oA0), wTM_g[pass], F(oA2), NTOK * NTOK, 128, 128,
              128, 1, Z0, Z0, 128, 1, Z0, Z0, 128, 1, Z0, Z0, BIG, 0, 1, 1, 1.f, 0);
        gemmL(stream, H(oA5), wTM_o[pass], F(oA4), NTOK * NTOK, 128, 128,
              128, 1, Z0, Z0, 128, 1, Z0, Z0, 128, 1, Z0, Z0, BIG, 0, 1, 1, 1.f, 0);
        msab_sig_gate_add<<<dim3((unsigned)((n8 + 255) / 256)), dim3(256), 0, stream>>>(
            F(oA2), F(oA4), out_z, n8);
    }

    // =============== triangle attention (start, then end/transposed) ===========
    for (int pass = 0; pass < 2; ++pass) {  // 0 = start, 1 = end
        int lng = pass ? TAE_LNG : TAS_LNG, lnb = pass ? TAE_LNB : TAS_LNB;
        ln(out_z, lng, lnb, H(oA0), NTOK * NTOK, CZ, CZ, 1, pass /*transposed rows*/);
        gemmL(stream, H(oA0), wTA_q[pass], H(oA3), NTOK * NTOK, 128, 128,
              128, 1, Z0, Z0, 128, 1, Z0, Z0, 128, 1, Z0, Z0, BIG, 0, 1, 1, 1.f, 1);
        gemmL(stream, H(oA0), wTA_k[pass], H(oA4), NTOK * NTOK, 128, 128,
              128, 1, Z0, Z0, 128, 1, Z0, Z0, 128, 1, Z0, Z0, BIG, 0, 1, 1, 1.f, 1);
        gemmL(stream, H(oA0), wTA_v[pass], H(oA5), NTOK * NTOK, 128, 128,
              128, 1, Z0, Z0, 128, 1, Z0, Z0, 128, 1, Z0, Z0, BIG, 0, 1, 1, 1.f, 1);
        gemmL(stream, H(oA0), wTA_g[pass], F(oA1), NTOK * NTOK, 128, 128,
              128, 1, Z0, Z0, 128, 1, Z0, Z0, 128, 1, Z0, Z0, BIG, 0, 1, 1, 1.f, 0);
        gemmL(stream, H(oA0), wTA_bias[pass], F(oA6), NTOK * NTOK, 4, 128,
              128, 1, Z0, Z0, 128, 1, Z0, Z0, 4, 1, Z0, Z0, BIG, 0, 1, 1, 1.f, 0);
        const float scale = 0.1767766953f;  // 1/sqrt(32)
        long ms0 = pass ? 1 : NTOK, ms1 = pass ? NTOK : 1;
        for (int a0 = 0; a0 < NTOK; a0 += 32) {
            // logits chunk f32 [a_l][h][b][b'], batch (a_l, h)
            gemmL(stream, H(oA3) + (size_t)a0 * 32768, H(oA4) + (size_t)a0 * 32768,
                  F(oA2), 256, 256, 32,
                  128, 1, 32768, 32,
                  128, 1, 32768, 32,
                  256, 1, 262144, 65536, BIG, 0, 32, 4, 1.f, 0);
            msab_tri_softmax<<<dim3(128), dim3(256), 0, stream>>>(
                F(oA2), F(oA6), mask, H(oA0), a0, ms0, ms1, scale);
            // o[a_l][b][h][c] = p @ v  (batch (a_l, h); B strided as v^T)
            gemmL(stream, H(oA0), H(oA5) + (size_t)a0 * 32768, F(oA2),
                  256, 32, 256,
                  256, 1, 262144, 65536,
                  1, 128, 32768, 32,
                  128, 1, 32768, 32, BIG, 0, 32, 4, 1.f, 0);
            long n = 32L * 256 * 128;
            msab_att_gate<<<dim3((unsigned)((n + 255) / 256)), dim3(256), 0, stream>>>(
                F(oA1), F(oA2), H(oA6 + 2 * MB), a0);
            if (!pass) {  // start: z[(a0+a_l), b] += delta
                gemmL(stream, H(oA6 + 2 * MB), wTA_o[pass],
                      out_z + (size_t)a0 * 32768, 8192, 128, 128,
                      128, 1, Z0, Z0, 128, 1, Z0, Z0,
                      128, 1, Z0, Z0, BIG, 0, 1, 1, 1.f, 2);
            } else {      // end: z[b, (a0+a_l)] += delta  (transposed row write)
                gemmL(stream, H(oA6 + 2 * MB), wTA_o[pass],
                      out_z + (size_t)a0 * 128, 8192, 128, 128,
                      128, 1, Z0, Z0, 128, 1, Z0, Z0,
                      32768, 1, Z0, Z0, 256, 128, 1, 1, 1.f, 2);
            }
        }
    }

    // =============== pair transition (SwiGLU, n=4), chunked ===============
    for (int r0 = 0; r0 < NTOK * NTOK; r0 += 16384) {
        float* xin = out_z + (size_t)r0 * CZ;
        ln(xin, PT_LNG, PT_LNB, H(oA0), 16384, CZ, CZ, 1, 0);
        gemmL(stream, H(oA0), wPTa, H(oA3), 16384, 512, 128,
              128, 1, Z0, Z0, 128, 1, Z0, Z0, 512, 1, Z0, Z0, BIG, 0, 1, 1, 1.f, 1);
        gemmL(stream, H(oA0), wPTb, H(oA4), 16384, 512, 128,
              128, 1, Z0, Z0, 128, 1, Z0, Z0, 512, 1, Z0, Z0, BIG, 0, 1, 1, 1.f, 1);
        long n = 16384L * 512;
        msab_silu_mul<<<dim3((unsigned)((n + 255) / 256)), dim3(256), 0, stream>>>(
            H(oA3), H(oA4), H(oA5), n);
        gemmL(stream, H(oA5), wPTo, xin, 16384, 128, 512,
              512, 1, Z0, Z0, 512, 1, Z0, Z0, 128, 1, Z0, Z0, BIG, 0, 1, 1, 1.f, 2);
    }
}
